// NRIModel_8787503088035
// MI455X (gfx1250) — matrix operand order repaired and checked by reading, not compiled, not run on hardware
//
#include <hip/hip_runtime.h>
#include <hip/hip_bf16.h>
#include <math.h>

// ---------------------------------------------------------------------------
// NRI model for MI455X (gfx1250): bf16 WMMA everywhere the math is a GEMM.
// Decoder message-passing (155 GFLOP of the ~165 GFLOP total) is fused:
// one block per (b,t,receiver-node) runs both MLP layers for its 31 edges
// as 16x16x32 bf16 WMMA tiles, weights the result by rel_type, and reduces
// over edges in-registers (edges are receiver-sorted, so no atomics).
// ---------------------------------------------------------------------------

typedef __bf16 bf16_t;
typedef __attribute__((ext_vector_type(16))) __bf16 v16bf;
typedef __attribute__((ext_vector_type(8)))  __bf16 v8bf;
typedef __attribute__((ext_vector_type(8)))  float  v8f;

#define DEV __device__ __forceinline__

// Model constants
#define Bc   8
#define Nn   32
#define Tt   49
#define Dd   4
#define Ee   992            // N*(N-1)
#define HEc  256
#define ME   (Bc*Ee)        // 7936  edge rows
#define MN   (Bc*Nn)        // 256   node rows
#define MBTN (Bc*Tt*Nn)     // 12544 (b,t,n) rows
#define OUT0 (Bc*Nn*48*Dd)  // 49152 floats of "output"

// ---- WMMA fragment helpers (contiguous-k order, same for A and B) ---------------------------
// A (16x32 bf16): lane L holds row m=L%16; K = [16g .. 16g+15] contiguous
DEV v16bf ld_a_frag(const bf16_t* rowk, int g) {
  v8bf lo = *(const v8bf*)(rowk + 16 * g);
  v8bf hi = *(const v8bf*)(rowk + 16 * g + 8);
  v16bf r;
#pragma unroll
  for (int i = 0; i < 8; ++i) { r[i] = lo[i]; r[i + 8] = hi[i]; }
  return r;
}
// B (32x16 bf16) from W^T (N x K row-major): lane L holds col n=L%16,
// K = [16g .. 16g+15] contiguous.
DEV v16bf ld_b_frag(const bf16_t* rowk, int g) {
  return *(const v16bf*)(rowk + 16 * g);
}
DEV v8f zero8() { v8f z; 
#pragma unroll
  for (int i = 0; i < 8; ++i) z[i] = 0.f; return z; }

DEV v8f wmma_bf16(v16bf a, v16bf b, v8f c) {
  return __builtin_amdgcn_wmma_f32_16x16x32_bf16(false, a, false, b,
                                                 (short)0, c, false, false);
}

// ---- Generic 64x64-tile GEMM:  C = act(A(MxK) @ W(KxN) + bias) ------------
// A: bf16 row-major (K contiguous, K%32==0).  WT: bf16, N x K row-major.
// ACT: 0 none, 1 ELU, 2 ReLU.  WF/WB: write fp32 / bf16 outputs.
template <int ACT, int WF, int WB>
__global__ __launch_bounds__(128) void k_gemm(
    const bf16_t* __restrict__ A, const bf16_t* __restrict__ WT,
    const float* __restrict__ bias, float* __restrict__ Cf,
    bf16_t* __restrict__ Cb, int M, int Kc, int Nc) {
  const int lane = threadIdx.x & 31;
  const int w    = threadIdx.x >> 5;
  const int g    = lane >> 4;
  const int l16  = lane & 15;
  const int ncol = blockIdx.y * 64 + w * 16 + l16;
  const int mb   = blockIdx.x * 64;
  const int nk   = Kc >> 5;
  v8f acc[4] = {zero8(), zero8(), zero8(), zero8()};
  const bf16_t* wrow = WT + (size_t)ncol * Kc;
  for (int kc = 0; kc < nk; ++kc) {
    v16bf bf_ = ld_b_frag(wrow + kc * 32, g);
#pragma unroll
    for (int mt = 0; mt < 4; ++mt) {
      const bf16_t* ap = A + (size_t)(mb + mt * 16 + l16) * Kc + kc * 32;
      acc[mt] = wmma_bf16(ld_a_frag(ap, g), bf_, acc[mt]);
    }
  }
  const float bb = bias[ncol];
#pragma unroll
  for (int mt = 0; mt < 4; ++mt) {
#pragma unroll
    for (int v = 0; v < 8; ++v) {
      int row = mb + mt * 16 + v + g * 8;   // C layout: vgpr v, lane-half g
      float x = acc[mt][v] + bb;
      if (ACT == 1) x = (x > 0.f) ? x : (__expf(x) - 1.f);
      if (ACT == 2) x = fmaxf(x, 0.f);
      size_t o = (size_t)row * Nc + ncol;
      if (WF) Cf[o] = x;
      if (WB) Cb[o] = (bf16_t)x;
    }
  }
}

// ---- Fused decoder message MLP + edge reduction ---------------------------
// grid = B*T*N blocks of 128 (4 waves). Block bid -> (b,t,n). Edges of node n
// are e = 31n+j, sender = j (+1 if j>=n). 2-layer MLP per k in {1,2,3},
// weighted by rel_type[b,e,k], reduced over the 31 edges.
__global__ __launch_bounds__(128) void k_decoder(
    const float* __restrict__ data, const float* __restrict__ rel,
    const bf16_t* __restrict__ mw1T, const bf16_t* __restrict__ mw2T,
    const float* __restrict__ mb1, const float* __restrict__ mb2,
    float* __restrict__ aggF) {
  __shared__ bf16_t lA[32 * 32];    // pre_msg tile, K padded 8->32
  __shared__ bf16_t lH[32 * 256];   // layer-1 activations
  __shared__ float  lW[32];         // per-edge rel_type weight
  const int tid = threadIdx.x;
  const int lane = tid & 31, w = tid >> 5;
  const int g = lane >> 4, l16 = lane & 15;
  const int n = blockIdx.x & 31;
  const int t = (blockIdx.x >> 5) % Tt;
  const int b = blockIdx.x / (Nn * Tt);

  for (int idx = tid; idx < 1024; idx += 128) {
    int m = idx >> 5, kk = idx & 31;
    float v = 0.f;
    if (m < 31 && kk < 8) {
      int snd  = (m < n) ? m : m + 1;
      int node = (kk < 4) ? snd : n;
      v = data[(((size_t)b * Nn + node) * Tt + t) * Dd + (kk & 3)];
    }
    lA[idx] = (bf16_t)v;
  }

  float aggP[4] = {0.f, 0.f, 0.f, 0.f};

  for (int k = 1; k < 4; ++k) {
    __syncthreads();
    if (tid < 32)
      lW[tid] = (tid < 31) ? rel[(((size_t)b * Ee) + n * 31 + tid) * 4 + k] : 0.f;
    __syncthreads();
    // ---- layer 1: 32x32 @ 32x256 -> relu -> lH -------------------------
#pragma unroll
    for (int i = 0; i < 4; ++i) {
      int ng = w * 4 + i;
      v16bf bf_ = ld_b_frag(mw1T + ((size_t)k * 256 + ng * 16 + l16) * 32, g);
      float b1 = mb1[k * 256 + ng * 16 + l16];
#pragma unroll
      for (int mt = 0; mt < 2; ++mt) {
        v16bf af = ld_a_frag(&lA[(mt * 16 + l16) * 32], g);
        v8f c = wmma_bf16(af, bf_, zero8());
#pragma unroll
        for (int v = 0; v < 8; ++v) {
          int row = mt * 16 + v + g * 8;
          lH[row * 256 + ng * 16 + l16] = (bf16_t)fmaxf(c[v] + b1, 0.f);
        }
      }
    }
    __syncthreads();
    // ---- layer 2: 32x256 @ 256x256 -------------------------------------
    v8f acc[4][2];
#pragma unroll
    for (int i = 0; i < 4; ++i) { acc[i][0] = zero8(); acc[i][1] = zero8(); }
    for (int kc = 0; kc < 8; ++kc) {
      v16bf a0 = ld_a_frag(&lH[(l16)      * 256 + kc * 32], g);
      v16bf a1 = ld_a_frag(&lH[(16 + l16) * 256 + kc * 32], g);
#pragma unroll
      for (int i = 0; i < 4; ++i) {
        int ng = w * 4 + i;
        v16bf bf_ = ld_b_frag(
            mw2T + (size_t)k * 65536 + (size_t)(ng * 16 + l16) * 256 + kc * 32, g);
        acc[i][0] = wmma_bf16(a0, bf_, acc[i][0]);
        acc[i][1] = wmma_bf16(a1, bf_, acc[i][1]);
      }
    }
    // ---- relu + rel_type weight + reduce over edges --------------------
#pragma unroll
    for (int i = 0; i < 4; ++i) {
      int ng = w * 4 + i;
      float b2 = mb2[k * 256 + ng * 16 + l16];
#pragma unroll
      for (int mt = 0; mt < 2; ++mt) {
#pragma unroll
        for (int v = 0; v < 8; ++v) {
          int row = mt * 16 + v + g * 8;
          aggP[i] += fmaxf(acc[i][mt][v] + b2, 0.f) * lW[row];
        }
      }
    }
  }
#pragma unroll
  for (int i = 0; i < 4; ++i) {
    float s = aggP[i] + __shfl_xor(aggP[i], 16, 32);  // combine lane halves
    if (lane < 16)
      aggF[(size_t)blockIdx.x * 256 + w * 64 + i * 16 + lane] = s;
  }
}

// ---- small VALU / prep kernels --------------------------------------------
__global__ __launch_bounds__(256) void k_wT(const float* __restrict__ W,
                                            bf16_t* __restrict__ WT, int Kin,
                                            int Kp, int Nout) {
  int i = blockIdx.x * 256 + threadIdx.x;
  if (i >= Nout * Kp) return;
  int nn = i / Kp, kk = i % Kp;
  WT[i] = (bf16_t)((kk < Kin) ? W[(size_t)kk * Nout + nn] : 0.f);
}

__global__ __launch_bounds__(256) void k_cvt_pad(const float* __restrict__ X,
                                                 bf16_t* __restrict__ Y, int M,
                                                 int Kin, int Kp) {
  int i = blockIdx.x * 256 + threadIdx.x;
  if (i >= M * Kp) return;
  int r = i / Kp, kk = i % Kp;
  Y[i] = (bf16_t)((kk < Kin) ? X[(size_t)r * Kin + kk] : 0.f);
}

__global__ __launch_bounds__(256) void k_stats(const float* __restrict__ X,
                                               int M, float* __restrict__ mean,
                                               float* __restrict__ inv) {
  __shared__ float s1[256], s2[256];
  int c = blockIdx.x;
  float a = 0.f, b = 0.f;
  for (int m = threadIdx.x; m < M; m += 256) {
    float v = X[(size_t)m * 256 + c];
    a += v; b += v * v;
  }
  s1[threadIdx.x] = a; s2[threadIdx.x] = b;
  __syncthreads();
  for (int off = 128; off > 0; off >>= 1) {
    if (threadIdx.x < off) {
      s1[threadIdx.x] += s1[threadIdx.x + off];
      s2[threadIdx.x] += s2[threadIdx.x + off];
    }
    __syncthreads();
  }
  if (threadIdx.x == 0) {
    float mu = s1[0] / M;
    mean[c] = mu;
    inv[c] = rsqrtf(s2[0] / M - mu * mu + 1e-5f);
  }
}

__global__ __launch_bounds__(256) void k_bn(const float* __restrict__ X,
    const float* __restrict__ mean, const float* __restrict__ inv,
    const float* __restrict__ gam, const float* __restrict__ bet,
    float* __restrict__ Yf, bf16_t* __restrict__ Yb, int total, int writeF) {
  int i = blockIdx.x * 256 + threadIdx.x;
  if (i >= total) return;
  int c = i & 255;
  float y = (X[i] - mean[c]) * inv[c] * gam[c] + bet[c];
  if (writeF) Yf[i] = y;
  Yb[i] = (bf16_t)y;
}

__global__ __launch_bounds__(256) void k_edges2(const bf16_t* __restrict__ xn,
    const int* __restrict__ sendI, const int* __restrict__ recvI,
    bf16_t* __restrict__ Y) {
  int i = blockIdx.x * 256 + threadIdx.x;
  if (i >= ME * 512) return;
  int r = i >> 9, c = i & 511;
  int b = r / Ee, e = r % Ee;
  int node = (c < 256) ? sendI[e] : recvI[e];
  Y[i] = xn[((size_t)b * Nn + node) * 256 + (c & 255)];
}

__global__ __launch_bounds__(256) void k_cat4(const bf16_t* __restrict__ x3n,
    const bf16_t* __restrict__ x2n, const int* __restrict__ sendI,
    const int* __restrict__ recvI, bf16_t* __restrict__ Y) {
  int i = blockIdx.x * 256 + threadIdx.x;
  if (i >= ME * 768) return;
  int r = i / 768, c = i % 768;
  int b = r / Ee, e = r % Ee;
  bf16_t v;
  if (c < 256)      v = x3n[((size_t)b * Nn + sendI[e]) * 256 + c];
  else if (c < 512) v = x3n[((size_t)b * Nn + recvI[e]) * 256 + (c - 256)];
  else              v = x2n[(size_t)r * 256 + (c - 512)];
  Y[i] = v;
}

__global__ __launch_bounds__(256) void k_incoming(const float* __restrict__ x2n,
                                                  bf16_t* __restrict__ Y) {
  int i = blockIdx.x * 256 + threadIdx.x;
  if (i >= MN * 256) return;
  int r = i >> 8, c = i & 255;
  int b = r / Nn, n = r % Nn;
  float s = 0.f;
  for (int j = 0; j < 31; ++j)
    s += x2n[((size_t)b * Ee + n * 31 + j) * 256 + c];
  Y[i] = (bf16_t)(s / (float)Nn);
}

__global__ __launch_bounds__(128) void k_fc(const float* __restrict__ X,
    const float* __restrict__ fw, const float* __restrict__ fb,
    const float* __restrict__ gum, float* __restrict__ rel,
    float* __restrict__ prob) {
  int r = blockIdx.x * 128 + threadIdx.x;
  if (r >= ME) return;
  float lg[4];
#pragma unroll
  for (int j = 0; j < 4; ++j) lg[j] = fb[j];
  for (int c = 0; c < 256; ++c) {
    float x = X[(size_t)r * 256 + c];
#pragma unroll
    for (int j = 0; j < 4; ++j) lg[j] += x * fw[c * 4 + j];
  }
  float mx = fmaxf(fmaxf(lg[0], lg[1]), fmaxf(lg[2], lg[3]));
  float e0[4], s = 0.f;
#pragma unroll
  for (int j = 0; j < 4; ++j) { e0[j] = __expf(lg[j] - mx); s += e0[j]; }
#pragma unroll
  for (int j = 0; j < 4; ++j) prob[(size_t)r * 4 + j] = e0[j] / s;
  float z[4];
#pragma unroll
  for (int j = 0; j < 4; ++j) z[j] = (lg[j] + gum[(size_t)r * 4 + j]) * 2.0f;
  float mz = fmaxf(fmaxf(z[0], z[1]), fmaxf(z[2], z[3]));
  float s2 = 0.f;
#pragma unroll
  for (int j = 0; j < 4; ++j) { z[j] = __expf(z[j] - mz); s2 += z[j]; }
#pragma unroll
  for (int j = 0; j < 4; ++j) rel[(size_t)r * 4 + j] = z[j] / s2;
}

__global__ __launch_bounds__(256) void k_aug(const float* __restrict__ data,
    const float* __restrict__ aggF, bf16_t* __restrict__ Y) {
  int i = blockIdx.x * 256 + threadIdx.x;
  if (i >= MBTN * 288) return;
  int r = i / 288, c = i % 288;
  int n = r & 31, t = (r >> 5) % Tt, b = r / (Nn * Tt);
  float v = 0.f;
  if (c < 4)        v = data[(((size_t)b * Nn + n) * Tt + t) * 4 + c];
  else if (c < 260) v = aggF[(size_t)r * 256 + (c - 4)];
  Y[i] = (bf16_t)v;
}

__global__ __launch_bounds__(256) void k_out3(const bf16_t* __restrict__ p2,
    const float* __restrict__ ow3, const float* __restrict__ ob3,
    const float* __restrict__ data, float* __restrict__ out) {
  int i = blockIdx.x * 256 + threadIdx.x;
  if (i >= MBTN * 4) return;
  int r = i >> 2, d = i & 3;
  float s = ob3[d];
  for (int c = 0; c < 256; ++c)
    s += (float)p2[(size_t)r * 256 + c] * ow3[c * 4 + d];
  int n = r & 31, t = (r >> 5) % Tt, b = r / (Nn * Tt);
  s += data[(((size_t)b * Nn + n) * Tt + t) * 4 + d];
  if (t < 48)
    out[(((size_t)b * Nn + n) * 48 + t) * 4 + d] = s;
}

// ---------------------------------------------------------------------------
extern "C" void kernel_launch(void* const* d_in, const int* in_sizes, int n_in,
                              void* d_out, int out_size, void* d_ws,
                              size_t ws_size, hipStream_t stream) {
  (void)in_sizes; (void)n_in; (void)out_size; (void)ws_size;
  const float* data   = (const float*)d_in[0];
  const float* gumbel = (const float*)d_in[1];
  const int*   recvI  = (const int*)d_in[2];
  const int*   sendI  = (const int*)d_in[3];
  const float *e1w1=(const float*)d_in[4],  *e1b1=(const float*)d_in[5],
              *e1w2=(const float*)d_in[6],  *e1b2=(const float*)d_in[7],
              *e1g =(const float*)d_in[8],  *e1be=(const float*)d_in[9];
  const float *e2w1=(const float*)d_in[10], *e2b1=(const float*)d_in[11],
              *e2w2=(const float*)d_in[12], *e2b2=(const float*)d_in[13],
              *e2g =(const float*)d_in[14], *e2be=(const float*)d_in[15];
  const float *e3w1=(const float*)d_in[16], *e3b1=(const float*)d_in[17],
              *e3w2=(const float*)d_in[18], *e3b2=(const float*)d_in[19],
              *e3g =(const float*)d_in[20], *e3be=(const float*)d_in[21];
  const float *e4w1=(const float*)d_in[22], *e4b1=(const float*)d_in[23],
              *e4w2=(const float*)d_in[24], *e4b2=(const float*)d_in[25],
              *e4g =(const float*)d_in[26], *e4be=(const float*)d_in[27];
  const float *fcw=(const float*)d_in[28], *fcb=(const float*)d_in[29];
  const float *mw1=(const float*)d_in[30], *mb1=(const float*)d_in[31];
  const float *mw2=(const float*)d_in[32], *mb2=(const float*)d_in[33];
  const float *ow1=(const float*)d_in[34], *ob1=(const float*)d_in[35];
  const float *ow2=(const float*)d_in[36], *ob2=(const float*)d_in[37];
  const float *ow3=(const float*)d_in[38], *ob3=(const float*)d_in[39];

  float* outP  = (float*)d_out;
  float* probP = outP + OUT0;

  char* wp = (char*)d_ws;
  auto alloc = [&](size_t bytes) -> void* {
    void* r = (void*)wp;
    wp += (bytes + 255) & ~(size_t)255;
    return r;
  };
  // bf16 weight copies (N x Kpad, row-major)
  bf16_t* wT_e1w1 = (bf16_t*)alloc((size_t)256*224*2);
  bf16_t* wT_e1w2 = (bf16_t*)alloc((size_t)256*256*2);
  bf16_t* wT_e2w1 = (bf16_t*)alloc((size_t)256*512*2);
  bf16_t* wT_e2w2 = (bf16_t*)alloc((size_t)256*256*2);
  bf16_t* wT_e3w1 = (bf16_t*)alloc((size_t)256*256*2);
  bf16_t* wT_e3w2 = (bf16_t*)alloc((size_t)256*256*2);
  bf16_t* wT_e4w1 = (bf16_t*)alloc((size_t)256*768*2);
  bf16_t* wT_e4w2 = (bf16_t*)alloc((size_t)256*256*2);
  bf16_t* wT_ow1  = (bf16_t*)alloc((size_t)256*288*2);
  bf16_t* wT_ow2  = (bf16_t*)alloc((size_t)256*256*2);
  bf16_t* wT_mw1  = (bf16_t*)alloc((size_t)4*256*32*2);
  bf16_t* wT_mw2  = (bf16_t*)alloc((size_t)4*256*256*2);
  // activations
  bf16_t* dataBF  = (bf16_t*)alloc((size_t)MN*224*2);
  bf16_t* t1b     = (bf16_t*)alloc((size_t)ME*256*2);
  bf16_t* x1nBF   = (bf16_t*)alloc((size_t)MN*256*2);
  bf16_t* edges2B = (bf16_t*)alloc((size_t)ME*512*2);
  bf16_t* x2nBF   = (bf16_t*)alloc((size_t)ME*256*2);
  bf16_t* incBF   = (bf16_t*)alloc((size_t)MN*256*2);
  bf16_t* x3nBF   = (bf16_t*)alloc((size_t)MN*256*2);
  bf16_t* cat4B   = (bf16_t*)alloc((size_t)ME*768*2);
  bf16_t* x4nBF   = (bf16_t*)alloc((size_t)ME*256*2);
  bf16_t* augBF   = (bf16_t*)alloc((size_t)MBTN*288*2);
  bf16_t* p1BF    = (bf16_t*)alloc((size_t)MBTN*256*2);
  bf16_t* p2BF    = (bf16_t*)alloc((size_t)MBTN*256*2);
  float*  t2f     = (float*)alloc((size_t)ME*256*4);
  float*  x2nF    = (float*)alloc((size_t)ME*256*4);
  float*  x4nF    = (float*)alloc((size_t)ME*256*4);
  float*  relT    = (float*)alloc((size_t)ME*4*4);
  float*  meanB   = (float*)alloc(256*4);
  float*  invB    = (float*)alloc(256*4);
  float*  aggF    = (float*)alloc((size_t)MBTN*256*4);

#define GRID1(cnt) dim3(((cnt) + 255) / 256)
  // ---- weight prep ----
  k_wT<<<GRID1(256*224), 256, 0, stream>>>(e1w1, wT_e1w1, 196, 224, 256);
  k_wT<<<GRID1(256*256), 256, 0, stream>>>(e1w2, wT_e1w2, 256, 256, 256);
  k_wT<<<GRID1(256*512), 256, 0, stream>>>(e2w1, wT_e2w1, 512, 512, 256);
  k_wT<<<GRID1(256*256), 256, 0, stream>>>(e2w2, wT_e2w2, 256, 256, 256);
  k_wT<<<GRID1(256*256), 256, 0, stream>>>(e3w1, wT_e3w1, 256, 256, 256);
  k_wT<<<GRID1(256*256), 256, 0, stream>>>(e3w2, wT_e3w2, 256, 256, 256);
  k_wT<<<GRID1(256*768), 256, 0, stream>>>(e4w1, wT_e4w1, 768, 768, 256);
  k_wT<<<GRID1(256*256), 256, 0, stream>>>(e4w2, wT_e4w2, 256, 256, 256);
  k_wT<<<GRID1(256*288), 256, 0, stream>>>(ow1, wT_ow1, 260, 288, 256);
  k_wT<<<GRID1(256*256), 256, 0, stream>>>(ow2, wT_ow2, 256, 256, 256);
  for (int k = 0; k < 4; ++k) {
    k_wT<<<GRID1(256*32), 256, 0, stream>>>(mw1 + (size_t)k*8*256,
                                            wT_mw1 + (size_t)k*256*32, 8, 32, 256);
    k_wT<<<GRID1(256*256), 256, 0, stream>>>(mw2 + (size_t)k*256*256,
                                             wT_mw2 + (size_t)k*256*256, 256, 256, 256);
  }
  // ---- encoder ----
  k_cvt_pad<<<GRID1(MN*224), 256, 0, stream>>>(data, dataBF, MN, 196, 224);
  k_gemm<1,0,1><<<dim3(MN/64,4),128,0,stream>>>(dataBF, wT_e1w1, e1b1, nullptr, t1b, MN, 224, 256);
  k_gemm<1,1,0><<<dim3(MN/64,4),128,0,stream>>>(t1b, wT_e1w2, e1b2, t2f, nullptr, MN, 256, 256);
  k_stats<<<256,256,0,stream>>>(t2f, MN, meanB, invB);
  k_bn<<<GRID1(MN*256),256,0,stream>>>(t2f, meanB, invB, e1g, e1be, t2f, x1nBF, MN*256, 0);

  k_edges2<<<GRID1(ME*512),256,0,stream>>>(x1nBF, sendI, recvI, edges2B);
  k_gemm<1,0,1><<<dim3(ME/64,4),128,0,stream>>>(edges2B, wT_e2w1, e2b1, nullptr, t1b, ME, 512, 256);
  k_gemm<1,1,0><<<dim3(ME/64,4),128,0,stream>>>(t1b, wT_e2w2, e2b2, t2f, nullptr, ME, 256, 256);
  k_stats<<<256,256,0,stream>>>(t2f, ME, meanB, invB);
  k_bn<<<GRID1(ME*256),256,0,stream>>>(t2f, meanB, invB, e2g, e2be, x2nF, x2nBF, ME*256, 1);

  k_incoming<<<GRID1(MN*256),256,0,stream>>>(x2nF, incBF);
  k_gemm<1,0,1><<<dim3(MN/64,4),128,0,stream>>>(incBF, wT_e3w1, e3b1, nullptr, t1b, MN, 256, 256);
  k_gemm<1,1,0><<<dim3(MN/64,4),128,0,stream>>>(t1b, wT_e3w2, e3b2, t2f, nullptr, MN, 256, 256);
  k_stats<<<256,256,0,stream>>>(t2f, MN, meanB, invB);
  k_bn<<<GRID1(MN*256),256,0,stream>>>(t2f, meanB, invB, e3g, e3be, t2f, x3nBF, MN*256, 0);

  k_cat4<<<GRID1(ME*768),256,0,stream>>>(x3nBF, x2nBF, sendI, recvI, cat4B);
  k_gemm<1,0,1><<<dim3(ME/64,4),128,0,stream>>>(cat4B, wT_e4w1, e4b1, nullptr, t1b, ME, 768, 256);
  k_gemm<1,1,0><<<dim3(ME/64,4),128,0,stream>>>(t1b, wT_e4w2, e4b2, t2f, nullptr, ME, 256, 256);
  k_stats<<<256,256,0,stream>>>(t2f, ME, meanB, invB);
  k_bn<<<GRID1(ME*256),256,0,stream>>>(t2f, meanB, invB, e4g, e4be, x4nF, x4nBF, ME*256, 1);

  k_fc<<<dim3((ME+127)/128),128,0,stream>>>(x4nF, fcw, fcb, gumbel, relT, probP);

  // ---- decoder ----
  k_decoder<<<dim3(MBTN),128,0,stream>>>(data, relT, wT_mw1, wT_mw2, mb1, mb2, aggF);
  k_aug<<<GRID1(MBTN*288),256,0,stream>>>(data, aggF, augBF);
  k_gemm<2,0,1><<<dim3(MBTN/64,4),128,0,stream>>>(augBF, wT_ow1, ob1, nullptr, p1BF, MBTN, 288, 256);
  k_gemm<2,0,1><<<dim3(MBTN/64,4),128,0,stream>>>(p1BF, wT_ow2, ob2, nullptr, p2BF, MBTN, 256, 256);
  k_out3<<<GRID1(MBTN*4),256,0,stream>>>(p2BF, ow3, ob3, data, outP);
#undef GRID1
}
